// LocalAttentionBlock_20160576487785
// MI455X (gfx1250) — matrix operand order repaired and checked by reading, not compiled, not run on hardware
//
#include <hip/hip_runtime.h>
#include <hip/hip_bf16.h>

#define BQ    2
#define SQ    4096
#define DIMQ  512
#define HEADSQ 8
#define HDQ   64
#define FFNQ  1536
#define WINQ  16
#define MTOT  (BQ * SQ)          // 8192 rows

typedef __attribute__((ext_vector_type(16))) __bf16 bf16x16;
typedef __attribute__((ext_vector_type(8)))  float  f32x8;

union FragU { bf16x16 v; uint4 u[2]; };

// ---------------------------------------------------------------------------
// LDS fragment loaders (wave32, V_WMMA_F32_16X16X32_BF16 operand layouts).
// Staged tiles are [128 rows][32 K] bf16 with row stride 80 B (padded: 20*r
// mod 64 banks -> conflict-free b128 reads).
// A (16x32 MxK): lanes 0-15 row=lane, K = 0..7 & 16..23;  lanes 16-31 K+8.
// B (32x16 KxN): lanes 0-15 col=lane, K = 0..7 & 16..23;  lanes 16-31 K+8.
// ---------------------------------------------------------------------------
#define ROWB 80
#define STAGE_MAT (128 * ROWB)   // 10240 B per staged matrix

__device__ __forceinline__ bf16x16 lds_a_frag(const char* base, int rowloc, int lane) {
  const int r  = lane & 15;
  const int kb = (lane >> 4) << 3;          // 0 or 8 (elements)
  const char* p = base + (size_t)(rowloc + r) * ROWB + kb * 2;
  FragU f;
  f.u[0] = *(const uint4*)(p);              // K = kb .. kb+7
  f.u[1] = *(const uint4*)(p + 32);         // K = kb+16 .. kb+23
  return f.v;
}

__device__ __forceinline__ bf16x16 lds_b_frag(const char* base, int colloc, int lane) {
  const int c  = lane & 15;
  const int kb = (lane >> 4) << 3;          // 0 or 8 (elements)
  const char* p = base + (size_t)(colloc + c) * ROWB + kb * 2;
  FragU f;
  f.u[0] = *(const uint4*)(p);              // K = kb .. kb+7
  f.u[1] = *(const uint4*)(p + 32);         // K = kb+16 .. kb+23
  return f.v;
}

// one async 16B chunk: global (64-bit vaddr) -> LDS (32-bit dst offset)
__device__ __forceinline__ void async_copy16(uint32_t lds_addr, const void* gaddr) {
  asm volatile("global_load_async_to_lds_b128 %0, %1, off"
               :: "v"(lds_addr), "v"((uint64_t)(uintptr_t)gaddr) : "memory");
}

// ---------------------------------------------------------------------------
// Unified double-buffered async-LDS GEMM.
//   C[M,N] = A[M,K](bf16) * Bt[N,K]^T(bf16)
// Workgroup: 256 thr = 8 waves (4 M x 2 N), tile 128x128, K-step 32.
// Wave tile 32x64 -> 8 WMMA / K-step (16 for MODE 2).
// MODE 1: Cf = resid + acc                    (f32)
// MODE 2: Cb = silu(A*B1t) * (A*B2t)          (bf16, dual-B)
// MODE 3: QKV split: col n -> buffer n>>9, column n&511 (f32)
// ---------------------------------------------------------------------------
__device__ __forceinline__ float silu_f(float v) { return v / (1.f + __expf(-v)); }

template<int MODE>
__global__ __launch_bounds__(256)
void gemm_async_wmma(const __hip_bfloat16* __restrict__ A,
                     const __hip_bfloat16* __restrict__ B1t,
                     const __hip_bfloat16* __restrict__ B2t,
                     float* __restrict__ Cf,
                     __hip_bfloat16* __restrict__ Cb,
                     const float* __restrict__ resid,
                     int N, int K) {
  constexpr int NMAT  = (MODE == 2) ? 3 : 2;
  constexpr int STAGE = NMAT * STAGE_MAT;
  constexpr int NIN   = 2 * NMAT;          // async instrs issued per wave per step
  __shared__ __align__(16) char smem[2 * STAGE];

  const int tid  = threadIdx.x;
  const int lane = tid & 31;
  const int wave = tid >> 5;
  const int rowT = blockIdx.y * 128;                 // tile origin in M
  const int colT = blockIdx.x * 128;                 // tile origin in N
  const int row0 = (wave >> 1) * 32;                 // wave origin inside tile
  const int col0 = (wave & 1) * 64;
  const uint32_t smem_base = (uint32_t)(uintptr_t)smem;

  // issue one K-step stage: 512 x 16B chunks per matrix, 2 per thread
  auto issue = [&](int bufsel, int k0) {
    const uint32_t sb = smem_base + bufsel * STAGE;
    #pragma unroll
    for (int j = 0; j < 2; ++j) {
      const int c = tid + j * 256;                   // chunk 0..511
      const int row = c >> 2, seg = c & 3;
      async_copy16(sb + row * ROWB + seg * 16,
                   A + (size_t)(rowT + row) * K + k0 + seg * 8);
    }
    #pragma unroll
    for (int j = 0; j < 2; ++j) {
      const int c = tid + j * 256;
      const int row = c >> 2, seg = c & 3;
      async_copy16(sb + STAGE_MAT + row * ROWB + seg * 16,
                   B1t + (size_t)(colT + row) * K + k0 + seg * 8);
    }
    if (MODE == 2) {
      #pragma unroll
      for (int j = 0; j < 2; ++j) {
        const int c = tid + j * 256;
        const int row = c >> 2, seg = c & 3;
        async_copy16(sb + 2 * STAGE_MAT + row * ROWB + seg * 16,
                     B2t + (size_t)(colT + row) * K + k0 + seg * 8);
      }
    }
  };

  f32x8 accP[2][4] = {};                             // A*B1
  f32x8 accQ[2][4] = {};                             // A*B2 (MODE 2)

  const int iters = K >> 5;
  issue(0, 0);
  for (int it = 0; it < iters; ++it) {
    const int cur = it & 1;
    const bool more = (it + 1) < iters;
    if (more) {
      issue(cur ^ 1, (it + 1) << 5);
      asm volatile("s_wait_asynccnt %0" :: "n"(NIN) : "memory");
    } else {
      asm volatile("s_wait_asynccnt 0" ::: "memory");
    }
    __syncthreads();                                 // all waves' stage ready

    const char* sb = smem + cur * STAGE;
    // Preload ALL fragments of this step into distinct registers so the
    // scheduler can clause the ds_load_b128s and drain them with partial
    // s_wait_dscnt against the WMMA burst.
    bf16x16 a0, a1, b[4];
    a0 = lds_a_frag(sb, row0, lane);
    a1 = lds_a_frag(sb, row0 + 16, lane);
    #pragma unroll
    for (int tn = 0; tn < 4; ++tn) b[tn] = lds_b_frag(sb + STAGE_MAT, col0 + tn * 16, lane);
    #pragma unroll
    for (int tn = 0; tn < 4; ++tn) {
      accP[0][tn] = __builtin_amdgcn_wmma_f32_16x16x32_bf16(false, a0, false, b[tn], (short)0, accP[0][tn], false, false);
      accP[1][tn] = __builtin_amdgcn_wmma_f32_16x16x32_bf16(false, a1, false, b[tn], (short)0, accP[1][tn], false, false);
    }
    if (MODE == 2) {
      bf16x16 b2[4];
      #pragma unroll
      for (int tn = 0; tn < 4; ++tn) b2[tn] = lds_b_frag(sb + 2 * STAGE_MAT, col0 + tn * 16, lane);
      #pragma unroll
      for (int tn = 0; tn < 4; ++tn) {
        accQ[0][tn] = __builtin_amdgcn_wmma_f32_16x16x32_bf16(false, a0, false, b2[tn], (short)0, accQ[0][tn], false, false);
        accQ[1][tn] = __builtin_amdgcn_wmma_f32_16x16x32_bf16(false, a1, false, b2[tn], (short)0, accQ[1][tn], false, false);
      }
    }
    if (more) __syncthreads();                       // protect buffer overwritten next step
  }

  // ---- epilogue ----
  const int nloc = lane & 15;
  const int mloc = (lane >> 4) << 3;
  #pragma unroll
  for (int tm = 0; tm < 2; ++tm) {
    #pragma unroll
    for (int tn = 0; tn < 4; ++tn) {
      #pragma unroll
      for (int i = 0; i < 8; ++i) {
        const int m = rowT + row0 + tm * 16 + mloc + i;
        const int n = colT + col0 + tn * 16 + nloc;
        const float v = accP[tm][tn][i];
        if (MODE == 1) {
          const size_t idx = (size_t)m * N + n;
          Cf[idx] = resid[idx] + v;
        } else if (MODE == 2) {
          Cb[(size_t)m * N + n] = __float2bfloat16(silu_f(v) * accQ[tm][tn][i]);
        } else {  // MODE 3: QKV split over three contiguous [MTOT,512] buffers
          Cf[(size_t)(n >> 9) * ((size_t)MTOT * DIMQ) + (size_t)m * DIMQ + (n & 511)] = v;
        }
      }
    }
  }
}

// ---------------------------------------------------------------------------
// RMSNorm (f32 in) -> bf16 out.  One wave per 512-element row.
// ---------------------------------------------------------------------------
__global__ __launch_bounds__(256)
void rmsnorm_bf16_kernel(const float* __restrict__ x, const float* __restrict__ w,
                         __hip_bfloat16* __restrict__ out) {
  const int lane = threadIdx.x & 31;
  const int row  = blockIdx.x * 8 + (threadIdx.x >> 5);
  const float4* xr = (const float4*)(x + (size_t)row * DIMQ);

  float4 v[4];
  float ss = 0.f;
  #pragma unroll
  for (int i = 0; i < 4; ++i) {
    v[i] = xr[lane + i * 32];
    ss += v[i].x * v[i].x + v[i].y * v[i].y + v[i].z * v[i].z + v[i].w * v[i].w;
  }
  #pragma unroll
  for (int off = 16; off > 0; off >>= 1) ss += __shfl_xor(ss, off, 32);
  const float scale = rsqrtf(ss * (1.0f / DIMQ) + 1e-6f);

  __hip_bfloat16* op = out + (size_t)row * DIMQ;
  #pragma unroll
  for (int i = 0; i < 4; ++i) {
    const int col = (lane + i * 32) * 4;
    op[col + 0] = __float2bfloat16(v[i].x * scale * w[col + 0]);
    op[col + 1] = __float2bfloat16(v[i].y * scale * w[col + 1]);
    op[col + 2] = __float2bfloat16(v[i].z * scale * w[col + 2]);
    op[col + 3] = __float2bfloat16(v[i].w * scale * w[col + 3]);
  }
}

// ---------------------------------------------------------------------------
// Transpose + convert: W [K][N] f32  ->  Wt [N][K] bf16
// ---------------------------------------------------------------------------
__global__ __launch_bounds__(256)
void transpose_to_bf16_kernel(const float* __restrict__ W, __hip_bfloat16* __restrict__ Wt,
                              int K, int N) {
  const int i = blockIdx.x * 256 + threadIdx.x;
  if (i >= K * N) return;
  const int n = i / K;
  const int k = i - n * K;
  Wt[i] = __float2bfloat16(W[(size_t)k * N + n]);
}

// ---------------------------------------------------------------------------
// RoPE in-place on Q,K (f32, layout [b*S+s, h*64+d]).
// ---------------------------------------------------------------------------
__global__ __launch_bounds__(256)
void rope_kernel(float* __restrict__ Q, float* __restrict__ K,
                 const float* __restrict__ inv_freq) {
  const int i = blockIdx.x * 256 + threadIdx.x;     // B*S*H*32 threads
  const int d  = i & 31;
  const int h  = (i >> 5) & 7;
  const int bs = i >> 8;                            // b*S + s
  const int s  = bs & (SQ - 1);
  const float ang = (float)s * inv_freq[d];
  float c, sn;
  __sincosf(ang, &sn, &c);
  const size_t i1 = (size_t)bs * DIMQ + h * HDQ + d;
  const size_t i2 = i1 + 32;
  const float q1 = Q[i1], q2 = Q[i2];
  Q[i1] = q1 * c - q2 * sn;
  Q[i2] = q2 * c + q1 * sn;
  const float k1 = K[i1], k2 = K[i2];
  K[i1] = k1 * c - k2 * sn;
  K[i2] = k2 * c + k1 * sn;
}

// ---------------------------------------------------------------------------
// Sliding-window attention.  Block = 64 threads handles 64 consecutive s for
// one (b,h).  K/V band (96 rows x 64 f32) staged in LDS (48 KB).
// Output bf16 at [b*S+s, h*64+d] -> directly the A matrix of the O-proj GEMM.
// ---------------------------------------------------------------------------
#define ATS 64
#define AROWS (ATS + 2 * WINQ)   // 96

__global__ __launch_bounds__(64)
void local_attn_kernel(const float* __restrict__ Q, const float* __restrict__ K,
                       const float* __restrict__ V,
                       const unsigned char* __restrict__ kpm,
                       __hip_bfloat16* __restrict__ out) {
  __shared__ float ks[AROWS * HDQ];
  __shared__ float vs[AROWS * HDQ];

  const int b  = blockIdx.z;
  const int h  = blockIdx.y;
  const int s0 = blockIdx.x * ATS;
  const int tid = threadIdx.x;

  for (int i = tid; i < AROWS * 16; i += ATS) {
    const int row = i >> 4;
    const int c4  = i & 15;
    int pos = s0 - WINQ + row;
    pos = pos < 0 ? 0 : (pos > SQ - 1 ? SQ - 1 : pos);
    const size_t base = ((size_t)(b * SQ + pos)) * DIMQ + h * HDQ;
    ((float4*)ks)[row * 16 + c4] = ((const float4*)(K + base))[c4];
    ((float4*)vs)[row * 16 + c4] = ((const float4*)(V + base))[c4];
  }
  __syncthreads();

  const int s = s0 + tid;
  const size_t qbase = ((size_t)(b * SQ + s)) * DIMQ + h * HDQ;
  float qv[HDQ];
  #pragma unroll
  for (int d = 0; d < HDQ; ++d) qv[d] = Q[qbase + d];

  float sc[2 * WINQ + 1];
  float mx = -1e30f;
  for (int w = 0; w < 2 * WINQ + 1; ++w) {
    const int pos = s + w - WINQ;
    const int posc = pos < 0 ? 0 : (pos > SQ - 1 ? SQ - 1 : pos);
    const bool keep = (pos >= 0) && (pos < SQ) && (kpm[(size_t)b * SQ + posc] == 0);
    const float* kr = ks + (tid + w) * HDQ;
    float dot = 0.f;
    #pragma unroll
    for (int d = 0; d < HDQ; ++d) dot += qv[d] * kr[d];
    sc[w] = keep ? dot * 0.125f : -1e9f;
    mx = fmaxf(mx, sc[w]);
  }
  float denom = 0.f;
  for (int w = 0; w < 2 * WINQ + 1; ++w) {
    sc[w] = __expf(sc[w] - mx);
    denom += sc[w];
  }
  float o[HDQ];
  #pragma unroll
  for (int d = 0; d < HDQ; ++d) o[d] = 0.f;
  for (int w = 0; w < 2 * WINQ + 1; ++w) {
    const float p = sc[w];
    const float* vr = vs + (tid + w) * HDQ;
    #pragma unroll
    for (int d = 0; d < HDQ; ++d) o[d] += p * vr[d];
  }
  const float inv = 1.f / denom;
  __hip_bfloat16* op = out + qbase;
  #pragma unroll
  for (int d = 0; d < HDQ; ++d) op[d] = __float2bfloat16(o[d] * inv);
}

// ---------------------------------------------------------------------------
// Host launcher
// ---------------------------------------------------------------------------
extern "C" void kernel_launch(void* const* d_in, const int* in_sizes, int n_in,
                              void* d_out, int out_size, void* d_ws, size_t ws_size,
                              hipStream_t stream) {
  const float* x        = (const float*)d_in[0];
  const float* inv_freq = (const float*)d_in[1];
  // d_in[2] position_ids == arange(S) broadcast; positions derived from s directly
  const unsigned char* kpm = (const unsigned char*)d_in[3];
  const float* norm1_w  = (const float*)d_in[4];
  const float* wq = (const float*)d_in[5];
  const float* wk = (const float*)d_in[6];
  const float* wv = (const float*)d_in[7];
  const float* wo = (const float*)d_in[8];
  const float* norm2_w  = (const float*)d_in[9];
  const float* w1 = (const float*)d_in[10];
  const float* w2 = (const float*)d_in[11];
  const float* w3 = (const float*)d_in[12];

  size_t off = 0;
  auto alloc = [&](size_t bytes) -> void* {
    void* p = (char*)d_ws + off;
    off += (bytes + 255) & ~(size_t)255;
    return p;
  };
  __hip_bfloat16* h1   = (__hip_bfloat16*)alloc((size_t)MTOT * DIMQ * 2);
  // wq/wk/wv transposed blocks are CONTIGUOUS -> one [1536][512] B matrix
  __hip_bfloat16* wqt  = (__hip_bfloat16*)alloc((size_t)DIMQ * DIMQ * 2);
  __hip_bfloat16* wkt  = (__hip_bfloat16*)alloc((size_t)DIMQ * DIMQ * 2);
  __hip_bfloat16* wvt  = (__hip_bfloat16*)alloc((size_t)DIMQ * DIMQ * 2);
  __hip_bfloat16* wot  = (__hip_bfloat16*)alloc((size_t)DIMQ * DIMQ * 2);
  __hip_bfloat16* w1t  = (__hip_bfloat16*)alloc((size_t)FFNQ * DIMQ * 2);
  __hip_bfloat16* w2t  = (__hip_bfloat16*)alloc((size_t)FFNQ * DIMQ * 2);
  __hip_bfloat16* w3t  = (__hip_bfloat16*)alloc((size_t)DIMQ * FFNQ * 2);
  // Q/K/V buffers contiguous -> MODE 3 split store
  float*          Qf   = (float*)alloc((size_t)MTOT * DIMQ * 4);
  float*          Kf   = (float*)alloc((size_t)MTOT * DIMQ * 4);
  float*          Vf   = (float*)alloc((size_t)MTOT * DIMQ * 4);
  __hip_bfloat16* attn = (__hip_bfloat16*)alloc((size_t)MTOT * DIMQ * 2);
  float*          x1   = (float*)alloc((size_t)MTOT * DIMQ * 4);
  __hip_bfloat16* h2   = (__hip_bfloat16*)alloc((size_t)MTOT * DIMQ * 2);
  __hip_bfloat16* G    = (__hip_bfloat16*)alloc((size_t)MTOT * FFNQ * 2);
  (void)Kf; (void)Vf;

  // 1) h1 = rmsnorm(x, norm1_w)  [bf16]
  rmsnorm_bf16_kernel<<<MTOT / 8, 256, 0, stream>>>(x, norm1_w, h1);

  // 2) weight transpose+convert
  transpose_to_bf16_kernel<<<(DIMQ * DIMQ + 255) / 256, 256, 0, stream>>>(wq, wqt, DIMQ, DIMQ);
  transpose_to_bf16_kernel<<<(DIMQ * DIMQ + 255) / 256, 256, 0, stream>>>(wk, wkt, DIMQ, DIMQ);
  transpose_to_bf16_kernel<<<(DIMQ * DIMQ + 255) / 256, 256, 0, stream>>>(wv, wvt, DIMQ, DIMQ);
  transpose_to_bf16_kernel<<<(DIMQ * DIMQ + 255) / 256, 256, 0, stream>>>(wo, wot, DIMQ, DIMQ);
  transpose_to_bf16_kernel<<<(DIMQ * FFNQ + 255) / 256, 256, 0, stream>>>(w1, w1t, DIMQ, FFNQ);
  transpose_to_bf16_kernel<<<(DIMQ * FFNQ + 255) / 256, 256, 0, stream>>>(w2, w2t, DIMQ, FFNQ);
  transpose_to_bf16_kernel<<<(FFNQ * DIMQ + 255) / 256, 256, 0, stream>>>(w3, w3t, FFNQ, DIMQ);

  // 3) fused QKV projection: one N=1536 GEMM, split f32 stores into Qf/Kf/Vf
  gemm_async_wmma<3><<<dim3(3 * DIMQ / 128, MTOT / 128), 256, 0, stream>>>(
      h1, wqt, nullptr, Qf, nullptr, nullptr, 3 * DIMQ, DIMQ);

  // 4) RoPE in place on Q,K
  rope_kernel<<<(MTOT * HEADSQ * 32) / 256, 256, 0, stream>>>(Qf, Kf, inv_freq);

  // 5) sliding-window attention -> bf16 [M, DIM]
  local_attn_kernel<<<dim3(SQ / ATS, HEADSQ, BQ), ATS, 0, stream>>>(Qf, Kf, Vf, kpm, attn);

  // 6) x1 = x + attn @ wo
  gemm_async_wmma<1><<<dim3(DIMQ / 128, MTOT / 128), 256, 0, stream>>>(
      attn, wot, nullptr, x1, nullptr, x, DIMQ, DIMQ);

  // 7) h2 = rmsnorm(x1, norm2_w)
  rmsnorm_bf16_kernel<<<MTOT / 8, 256, 0, stream>>>(x1, norm2_w, h2);

  // 8) G = silu(h2@w1) * (h2@w2)  [bf16, fused dual-GEMM, dual-B async stage]
  gemm_async_wmma<2><<<dim3(FFNQ / 128, MTOT / 128), 256, 0, stream>>>(
      h2, w1t, w2t, nullptr, G, nullptr, FFNQ, DIMQ);

  // 9) out = x1 + G @ w3
  gemm_async_wmma<1><<<dim3(DIMQ / 128, MTOT / 128), 256, 0, stream>>>(
      G, w3t, nullptr, (float*)d_out, nullptr, x1, DIMQ, FFNQ);
}
